// MHAModule_35734127903556
// MI455X (gfx1250) — matrix operand order repaired and checked by reading, not compiled, not run on hardware
//
#include <hip/hip_runtime.h>
#include <hip/hip_bf16.h>
#include <stdint.h>

// ---------------- problem constants ----------------
static constexpr int S_  = 192;
static constexpr int B_  = 192;
static constexpr int NH  = 8;
static constexpr int DH  = 64;
static constexpr int DM  = 512;
static constexpr float SCALE = 0.125f;          // 1/sqrt(64)
static constexpr size_t ROWS = (size_t)S_ * B_; // 36864

// ---------------- workspace layout (bytes) ----------------
static constexpr size_t OFF_H    = 0;                         // h (LN output) bf16 [rows][512]
static constexpr size_t SZ_H     = ROWS * DM * 2;
static constexpr size_t OFF_WQKV = OFF_H + SZ_H;              // W_qkv^T bf16 [512][1536]
static constexpr size_t SZ_WQKV  = (size_t)DM * (3 * DM) * 2;
static constexpr size_t OFF_WO   = OFF_WQKV + SZ_WQKV;        // W_o^T bf16 [512][512]
static constexpr size_t SZ_WO    = (size_t)DM * DM * 2;
static constexpr size_t OFF_PE   = OFF_WO + SZ_WO;            // pos_emb f32 [192][512]
static constexpr size_t SZ_PE    = (size_t)B_ * DM * 4;
static constexpr size_t OFF_RT   = OFF_PE + SZ_PE;            // r_head_k^T bf16 [8][64][192]
static constexpr size_t SZ_RT    = (size_t)NH * DH * S_ * 2;
static constexpr size_t OFF_QW   = OFF_RT + SZ_RT;            // q + r_w_bias bf16 [b][h][s][d]
static constexpr size_t SZ_Q     = (size_t)B_ * NH * S_ * DH * 2;
static constexpr size_t OFF_QR   = OFF_QW + SZ_Q;             // q + r_r_bias bf16 [b][h][s][d]
static constexpr size_t OFF_KT   = OFF_QR + SZ_Q;             // k^T bf16 [b][h][d][s]
static constexpr size_t OFF_V    = OFF_KT + SZ_Q;             // v bf16 [b][h][s][d]
static constexpr size_t OFF_AV   = OFF_V + SZ_Q;              // attn_vec bf16 [s][b][512]

// ---------------- WMMA helpers (CDNA5 gfx1250) ----------------
typedef __attribute__((ext_vector_type(16))) __bf16   bf16x16;
typedef __attribute__((ext_vector_type(8)))  float    f32x8;
typedef __attribute__((ext_vector_type(8)))  uint32_t u32x8;

__device__ __forceinline__ uint16_t f2bf(float f) {
  uint32_t u = __builtin_bit_cast(uint32_t, f);
  uint32_t r = (u + 0x7FFFu + ((u >> 16) & 1u)) >> 16;   // round-to-nearest-even
  return (uint16_t)r;
}

// A fragment: 16x32 bf16, row-major source (K contiguous), per ISA 7.12.2:
// lane = h*16+m ; VGPR v holds K = (v<4 ? 2v : 16+2(v-4)) + 8h  (pair)
// -> per lane this is two contiguous 16B chunks; compiler fuses into global_load_b128.
__device__ __forceinline__ bf16x16 load_a(const uint16_t* base, int ld, int row, int k0, int h) {
  const uint32_t* rp = (const uint32_t*)(base + (size_t)row * ld + k0);
  u32x8 r;
#pragma unroll
  for (int v = 0; v < 8; ++v) {
    int kb = ((v & 4) ? (16 + 2 * (v & 3)) : (2 * (v & 3))) + 8 * h;
    r[v] = rp[kb >> 1];
  }
  return __builtin_bit_cast(bf16x16, r);
}

// B fragment: 32x16 bf16, K-major source (N contiguous):
// lane = h*16+n ; VGPR v holds K = (v<4 ? 2v : 16+2(v-4)) + 8h  (pair), same as A
__device__ __forceinline__ bf16x16 load_b(const uint16_t* base, int ld, int k0, int n0, int lane) {
  int hh = lane >> 4;
  const uint16_t* cp = base + (size_t)k0 * ld + n0 + (lane & 15);
  u32x8 r;
#pragma unroll
  for (int v = 0; v < 8; ++v) {
    int kb = ((v & 4) ? (16 + 2 * (v & 3)) : (2 * (v & 3))) + 8 * hh;
    uint32_t lo = cp[(size_t)kb * ld], hi = cp[(size_t)(kb + 1) * ld];
    r[v] = lo | (hi << 16);
  }
  return __builtin_bit_cast(bf16x16, r);
}

__device__ __forceinline__ f32x8 wmma_bf16(bf16x16 a, bf16x16 b, f32x8 c) {
  return __builtin_amdgcn_wmma_f32_16x16x32_bf16(false, a, false, b, (short)0, c, false, false);
}

__device__ __forceinline__ f32x8 zero8() {
  f32x8 z = {0.f, 0.f, 0.f, 0.f, 0.f, 0.f, 0.f, 0.f};
  return z;
}

// ---------------- K0: weight transpose + bf16 cast ----------------
__global__ __launch_bounds__(256) void k_prep_w(const float* __restrict__ Wqkv,
                                                const float* __restrict__ Wo,
                                                uint16_t* __restrict__ WqkvT,
                                                uint16_t* __restrict__ WoT) {
  int idx = blockIdx.x * 256 + threadIdx.x;
  if (idx < DM * 3 * DM) {                       // WqkvT[d][e] = Wqkv[e][d]
    int d = idx / (3 * DM), e = idx % (3 * DM);
    WqkvT[idx] = f2bf(Wqkv[(size_t)e * DM + d]);
  }
  int i2 = idx - DM * 3 * DM;
  if (i2 >= 0 && i2 < DM * DM) {                 // WoT[d][e] = Wo[e][d]
    int d = i2 >> 9, e = i2 & 511;
    WoT[i2] = f2bf(Wo[(size_t)e * DM + d]);
  }
}

// ---------------- K1: fused double layernorm -> bf16 ----------------
__device__ __forceinline__ float block_sum(float v, float* red, int t) {
  red[t] = v;
  __syncthreads();
  for (int off = 128; off > 0; off >>= 1) {
    if (t < off) red[t] += red[t + off];
    __syncthreads();
  }
  float r = red[0];
  __syncthreads();
  return r;
}

__global__ __launch_bounds__(256) void k_ln(const float* __restrict__ x,
                                            const float* __restrict__ g1, const float* __restrict__ b1,
                                            const float* __restrict__ g2, const float* __restrict__ b2,
                                            uint16_t* __restrict__ hbuf) {
  __shared__ float red[256];
  int row = blockIdx.x, t = threadIdx.x;
  const float* xr = x + (size_t)row * DM;
  float v0 = xr[t], v1 = xr[t + 256];

  float mu = block_sum(v0 + v1, red, t) * (1.0f / 512.0f);
  float d0 = v0 - mu, d1 = v1 - mu;
  float rs = rsqrtf(block_sum(d0 * d0 + d1 * d1, red, t) * (1.0f / 512.0f) + 1e-5f);
  float y0 = d0 * rs * g1[t] + b1[t];
  float y1 = d1 * rs * g1[t + 256] + b1[t + 256];

  float mu2 = block_sum(y0 + y1, red, t) * (1.0f / 512.0f);
  float e0 = y0 - mu2, e1 = y1 - mu2;
  float rs2 = rsqrtf(block_sum(e0 * e0 + e1 * e1, red, t) * (1.0f / 512.0f) + 1e-5f);
  float z0 = e0 * rs2 * g2[t] + b2[t];
  float z1 = e1 * rs2 * g2[t + 256] + b2[t + 256];

  uint16_t* hr = hbuf + (size_t)row * DM;
  hr[t] = f2bf(z0);
  hr[t + 256] = f2bf(z1);
}

// ---------------- K2a: positional embedding ----------------
__global__ __launch_bounds__(256) void k_posemb(const float* __restrict__ inp, float* __restrict__ pe) {
  int idx = blockIdx.x * 256 + threadIdx.x;
  if (idx >= B_ * DM) return;
  int j = idx >> 9, c = idx & 511;
  float p = inp[(size_t)j * DM];                 // inputs[0, j, 0]
  int i2 = (c < 256) ? c : (c - 256);
  float invf = __expf(-((float)(2 * i2) / 512.0f) * 9.210340371976184f); // ln(10000)
  float a = p * invf;
  pe[idx] = (c < 256) ? __sinf(a) : __cosf(a);
}

// ---------------- K2b: r_head_k = pos_emb @ W_r^T  (stored [head][d][j] bf16) ----------------
__global__ __launch_bounds__(256) void k_rhead(const float* __restrict__ pe,
                                               const float* __restrict__ Wr,
                                               uint16_t* __restrict__ rT) {
  int idx = blockIdx.x * 256 + threadIdx.x;      // e * 192 + j, e = head*64+d
  if (idx >= NH * DH * S_) return;
  int j = idx % S_, e = idx / S_;
  const float* per = pe + (size_t)j * DM;
  const float* wr = Wr + (size_t)e * DM;
  float acc = 0.f;
  for (int c = 0; c < DM; ++c) acc += per[c] * wr[c];
  rT[(size_t)e * S_ + j] = f2bf(acc);
}

// ---------------- K3: QKV GEMM (WMMA, software-pipelined) + bias/layout epilogue ----------------
__global__ __launch_bounds__(256) void k_qkv(const uint16_t* __restrict__ hbuf,
                                             const uint16_t* __restrict__ Wt,
                                             const float* __restrict__ rwb,
                                             const float* __restrict__ rrb,
                                             uint16_t* __restrict__ qw, uint16_t* __restrict__ qr,
                                             uint16_t* __restrict__ kT, uint16_t* __restrict__ vv) {
  int wave = (blockIdx.x * 256 + threadIdx.x) >> 5;
  int lane = threadIdx.x & 31;
  int h = lane >> 4, ml = lane & 15;
  int m0 = (wave / 24) * 16;                     // 2304 M-tiles
  int n0 = (wave % 24) * 64;                     // 24 N-groups of 64
  f32x8 acc[4] = {zero8(), zero8(), zero8(), zero8()};

  constexpr int KSTEPS = DM / 32;
  bf16x16 A = load_a(hbuf, DM, m0 + ml, 0, h);
  bf16x16 Bc[4];
#pragma unroll
  for (int t = 0; t < 4; ++t) Bc[t] = load_b(Wt, 3 * DM, 0, n0 + 16 * t, lane);

#pragma unroll
  for (int kt = 0; kt < KSTEPS; ++kt) {          // fully unrolled: rotation = SSA rename
    int kn = (kt + 1 < KSTEPS) ? (kt + 1) * 32 : kt * 32;  // prefetch next (clamped on last)
    bf16x16 An = load_a(hbuf, DM, m0 + ml, kn, h);
    bf16x16 Bn[4];
#pragma unroll
    for (int t = 0; t < 4; ++t) Bn[t] = load_b(Wt, 3 * DM, kn, n0 + 16 * t, lane);
#pragma unroll
    for (int t = 0; t < 4; ++t) acc[t] = wmma_bf16(A, Bc[t], acc[t]);
    A = An;
#pragma unroll
    for (int t = 0; t < 4; ++t) Bc[t] = Bn[t];
  }
#pragma unroll
  for (int t = 0; t < 4; ++t) {
#pragma unroll
    for (int g = 0; g < 8; ++g) {
      int r = m0 + g + 8 * h;                    // row = s*B + b
      int e = n0 + 16 * t + ml;                  // 0..1535
      float val = acc[t][g];
      int s = r / B_, b = r % B_;
      int sect = e >> 9, ei = e & 511;
      int head = ei >> 6, d = ei & 63;
      size_t bh = (size_t)b * NH + head;
      if (sect == 0) {
        qw[(bh * S_ + s) * DH + d] = f2bf(val + rwb[ei]);
        qr[(bh * S_ + s) * DH + d] = f2bf(val + rrb[ei]);
      } else if (sect == 1) {
        kT[(bh * DH + d) * S_ + s] = f2bf(val);  // K stored d-major for B-fragments
      } else {
        vv[(bh * S_ + s) * DH + d] = f2bf(val);  // V stored j-major for B-fragments
      }
    }
  }
}

// ---------------- K4: fused attention per (b, head) ----------------
// scores in LDS (fp32), rel-shift as bijective scatter-add, softmax stats, P@V with WMMA
__global__ __launch_bounds__(384) void k_attn(const uint16_t* __restrict__ qw,
                                              const uint16_t* __restrict__ qr,
                                              const uint16_t* __restrict__ kT,
                                              const uint16_t* __restrict__ vv,
                                              const uint16_t* __restrict__ rT,
                                              uint16_t* __restrict__ av) {
  __shared__ float sc[S_ * S_];                  // 147456 B (CDNA5: 320KB/WG available)
  __shared__ float rowm[S_], rowr[S_];
  int bh = blockIdx.x;
  int b = bh >> 3, head = bh & 7;
  int t = threadIdx.x;
  int wave = t >> 5, lane = t & 31;
  int h = lane >> 4, ml = lane & 15;
  int i0 = wave * 16;                            // 12 waves x 16 rows = 192

  const uint16_t* qwb = qw + (size_t)bh * (S_ * DH);
  const uint16_t* qrb = qr + (size_t)bh * (S_ * DH);
  const uint16_t* kb  = kT + (size_t)bh * (DH * S_);
  const uint16_t* vb  = vv + (size_t)bh * (S_ * DH);
  const uint16_t* rb  = rT + (size_t)head * (DH * S_);

  // Phase A: AC = (q + r_w_bias) @ K^T  -> scores   (B fragments pipelined across tiles)
  {
    bf16x16 A0 = load_a(qwb, DH, i0 + ml, 0, h);
    bf16x16 A1 = load_a(qwb, DH, i0 + ml, 32, h);
    bf16x16 Bc0 = load_b(kb, S_, 0, 0, lane);
    bf16x16 Bc1 = load_b(kb, S_, 32, 0, lane);
#pragma unroll
    for (int jt = 0; jt < 12; ++jt) {
      int j0 = jt * 16;
      int jn = (jt < 11) ? j0 + 16 : j0;
      bf16x16 Bn0 = load_b(kb, S_, 0, jn, lane);
      bf16x16 Bn1 = load_b(kb, S_, 32, jn, lane);
      f32x8 acc = zero8();
      acc = wmma_bf16(A0, Bc0, acc);
      acc = wmma_bf16(A1, Bc1, acc);
#pragma unroll
      for (int g = 0; g < 8; ++g)
        sc[(i0 + g + 8 * h) * S_ + (j0 + ml)] = acc[g];
      Bc0 = Bn0;
      Bc1 = Bn1;
    }
  }
  __syncthreads();

  // Phase B: BD = (q + r_r_bias) @ r_head_k^T, rel-shifted scatter-add into scores.
  // out flat f = (i+1)*S + j ; source (si,sj) maps via f = si*(S+1) + sj + 1 (bijective)
  {
    bf16x16 A0 = load_a(qrb, DH, i0 + ml, 0, h);
    bf16x16 A1 = load_a(qrb, DH, i0 + ml, 32, h);
    bf16x16 Bc0 = load_b(rb, S_, 0, 0, lane);
    bf16x16 Bc1 = load_b(rb, S_, 32, 0, lane);
#pragma unroll
    for (int jt = 0; jt < 12; ++jt) {
      int j0 = jt * 16;
      int jn = (jt < 11) ? j0 + 16 : j0;
      bf16x16 Bn0 = load_b(rb, S_, 0, jn, lane);
      bf16x16 Bn1 = load_b(rb, S_, 32, jn, lane);
      f32x8 acc = zero8();
      acc = wmma_bf16(A0, Bc0, acc);
      acc = wmma_bf16(A1, Bc1, acc);
#pragma unroll
      for (int g = 0; g < 8; ++g) {
        int si = i0 + g + 8 * h;
        int sj = j0 + ml;
        int f = si * (S_ + 1) + sj + 1;
        int di = f / S_ - 1;
        int dj = f % S_;
        if (di >= 0) sc[di * S_ + dj] += acc[g]; // unique writer per element
      }
      Bc0 = Bn0;
      Bc1 = Bn1;
    }
  }
  __syncthreads();

  // Phase C: per-row softmax stats (max, 1/sum) with scaling
  if (t < S_) {
    const float* r = &sc[t * S_];
    float m = -3.4e38f;
    for (int j = 0; j < S_; ++j) m = fmaxf(m, r[j] * SCALE);
    float ssum = 0.f;
    for (int j = 0; j < S_; ++j) ssum += __expf(r[j] * SCALE - m);
    rowm[t] = m;
    rowr[t] = 1.0f / ssum;
  }
  __syncthreads();

  // Phase D: attn_vec = P @ V ; P-fragments rebuilt on the fly from LDS scores while
  // next V fragments are in flight (VALU/DS overlapping VMEM).
  f32x8 acc[4] = {zero8(), zero8(), zero8(), zero8()};
  {
    int row = i0 + ml;
    float m = rowm[row], rr = rowr[row];
    const float* srow = &sc[row * S_];
    bf16x16 Bc[4];
#pragma unroll
    for (int tt = 0; tt < 4; ++tt) Bc[tt] = load_b(vb, DH, 0, tt * 16, lane);
#pragma unroll
    for (int kt = 0; kt < 6; ++kt) {
      int j0 = kt * 32;
      int jn = (kt < 5) ? j0 + 32 : j0;
      bf16x16 Bn[4];
#pragma unroll
      for (int tt = 0; tt < 4; ++tt) Bn[tt] = load_b(vb, DH, jn, tt * 16, lane);
      u32x8 ar;
#pragma unroll
      for (int v = 0; v < 8; ++v) {
        int kb2 = ((v & 4) ? (16 + 2 * (v & 3)) : (2 * (v & 3))) + 8 * h;
        int j = j0 + kb2;
        float p0 = __expf(srow[j] * SCALE - m) * rr;
        float p1 = __expf(srow[j + 1] * SCALE - m) * rr;
        ar[v] = ((uint32_t)f2bf(p1) << 16) | (uint32_t)f2bf(p0);
      }
      bf16x16 A = __builtin_bit_cast(bf16x16, ar);
#pragma unroll
      for (int tt = 0; tt < 4; ++tt) acc[tt] = wmma_bf16(A, Bc[tt], acc[tt]);
#pragma unroll
      for (int tt = 0; tt < 4; ++tt) Bc[tt] = Bn[tt];
    }
  }
  // epilogue: av[s][b][head*64 + d]
#pragma unroll
  for (int tt = 0; tt < 4; ++tt) {
#pragma unroll
    for (int g = 0; g < 8; ++g) {
      int i = i0 + g + 8 * h;
      int col = tt * 16 + ml;
      av[((size_t)i * B_ + b) * DM + head * DH + col] = f2bf(acc[tt][g]);
    }
  }
}

// ---------------- K5: output GEMM (WMMA, software-pipelined): out = av @ W_o^T ----------------
__global__ __launch_bounds__(256) void k_out(const uint16_t* __restrict__ av,
                                             const uint16_t* __restrict__ WoT,
                                             float* __restrict__ out) {
  int wave = (blockIdx.x * 256 + threadIdx.x) >> 5;
  int lane = threadIdx.x & 31;
  int h = lane >> 4, ml = lane & 15;
  int m0 = (wave >> 3) * 16;                     // 2304 M-tiles
  int n0 = (wave & 7) * 64;                      // 8 N-groups
  f32x8 acc[4] = {zero8(), zero8(), zero8(), zero8()};

  constexpr int KSTEPS = DM / 32;
  bf16x16 A = load_a(av, DM, m0 + ml, 0, h);
  bf16x16 Bc[4];
#pragma unroll
  for (int tt = 0; tt < 4; ++tt) Bc[tt] = load_b(WoT, DM, 0, n0 + 16 * tt, lane);

#pragma unroll
  for (int kt = 0; kt < KSTEPS; ++kt) {          // fully unrolled: rotation = SSA rename
    int kn = (kt + 1 < KSTEPS) ? (kt + 1) * 32 : kt * 32;
    bf16x16 An = load_a(av, DM, m0 + ml, kn, h);
    bf16x16 Bn[4];
#pragma unroll
    for (int tt = 0; tt < 4; ++tt) Bn[tt] = load_b(WoT, DM, kn, n0 + 16 * tt, lane);
#pragma unroll
    for (int tt = 0; tt < 4; ++tt) acc[tt] = wmma_bf16(A, Bc[tt], acc[tt]);
    A = An;
#pragma unroll
    for (int tt = 0; tt < 4; ++tt) Bc[tt] = Bn[tt];
  }
#pragma unroll
  for (int tt = 0; tt < 4; ++tt) {
#pragma unroll
    for (int g = 0; g < 8; ++g)
      out[(size_t)(m0 + g + 8 * h) * DM + n0 + 16 * tt + ml] = acc[tt][g];
  }
}

// ---------------- launcher ----------------
extern "C" void kernel_launch(void* const* d_in, const int* in_sizes, int n_in,
                              void* d_out, int out_size, void* d_ws, size_t ws_size,
                              hipStream_t stream) {
  (void)in_sizes; (void)n_in; (void)out_size; (void)ws_size;
  const float* inputs = (const float*)d_in[0];
  const float* ln1_g  = (const float*)d_in[1];
  const float* ln1_b  = (const float*)d_in[2];
  const float* ln2_g  = (const float*)d_in[3];
  const float* ln2_b  = (const float*)d_in[4];
  const float* Wqkv   = (const float*)d_in[5];
  const float* Wr     = (const float*)d_in[6];
  const float* Wo     = (const float*)d_in[7];
  const float* rwb    = (const float*)d_in[8];
  const float* rrb    = (const float*)d_in[9];
  float* out = (float*)d_out;

  char* ws = (char*)d_ws;
  uint16_t* hbuf  = (uint16_t*)(ws + OFF_H);
  uint16_t* WqkvT = (uint16_t*)(ws + OFF_WQKV);
  uint16_t* WoT   = (uint16_t*)(ws + OFF_WO);
  float*    pe    = (float*)(ws + OFF_PE);
  uint16_t* rT    = (uint16_t*)(ws + OFF_RT);
  uint16_t* qwbuf = (uint16_t*)(ws + OFF_QW);
  uint16_t* qrbuf = (uint16_t*)(ws + OFF_QR);
  uint16_t* kTbuf = (uint16_t*)(ws + OFF_KT);
  uint16_t* vbuf  = (uint16_t*)(ws + OFF_V);
  uint16_t* avbuf = (uint16_t*)(ws + OFF_AV);

  k_prep_w<<<(DM * 3 * DM + DM * DM + 255) / 256, 256, 0, stream>>>(Wqkv, Wo, WqkvT, WoT);
  k_ln<<<(int)ROWS, 256, 0, stream>>>(inputs, ln1_g, ln1_b, ln2_g, ln2_b, hbuf);
  k_posemb<<<(B_ * DM + 255) / 256, 256, 0, stream>>>(inputs, pe);
  k_rhead<<<(NH * DH * S_ + 255) / 256, 256, 0, stream>>>(pe, Wr, rT);
  k_qkv<<<(int)((ROWS / 16) * 24 / 8), 256, 0, stream>>>(hbuf, WqkvT, rwb, rrb,
                                                         qwbuf, qrbuf, kTbuf, vbuf);
  k_attn<<<B_ * NH, 384, 0, stream>>>(qwbuf, qrbuf, kTbuf, vbuf, rT, avbuf);
  k_out<<<(int)((ROWS / 16) * 8 / 8), 256, 0, stream>>>(avbuf, WoT, out);
}
